// HybridSolver_89189290869206
// MI455X (gfx1250) — matrix operand order repaired and checked by reading, not compiled, not run on hardware
//
#include <hip/hip_runtime.h>

typedef __attribute__((ext_vector_type(16))) _Float16 v16h;
typedef __attribute__((ext_vector_type(8)))  float    v8f;

#define NN    128
#define TILE  16
#define INV_HH 16384.0f          // 1/(H*H), exact power of two
#define HH_OVER_4 (1.0f/65536.0f)

// ---------------------------------------------------------------------------
// Rearrange w2 (fp32 OIHW [64][64][3][3]) into f16 WMMA A-fragments:
// layout [tap][kc][ocTile][lane(32)][e(16)] in contiguous k order (B uses the same)
// (lanes 0-15: K = e ; lanes 16-31: K = 16 + e), M = ocTile*16 + (lane&15).
// ---------------------------------------------------------------------------
__global__ __launch_bounds__(256) void prep_w2_kernel(
    const float* __restrict__ w2, _Float16* __restrict__ dst)
{
    int flat = blockIdx.x * 256 + threadIdx.x;        // 9*2*4*32*16 = 36864
    if (flat >= 36864) return;
    int e    = flat & 15;
    int lane = (flat >> 4) & 31;
    int ocT  = (flat >> 9) & 3;
    int kc   = (flat >> 11) & 1;
    int tap  = flat >> 12;                            // 0..8
    int oc = ocT * 16 + (lane & 15);
    int ic = kc * 32 + ((lane >> 4) ? 16 : 0) + e;
    dst[flat] = (_Float16)w2[(oc * 64 + ic) * 9 + tap];
}

// ---------------------------------------------------------------------------
// Jacobi sweep: out = u + (H^2/4) * (f + lap(u)), periodic.
// ---------------------------------------------------------------------------
__global__ __launch_bounds__(256) void jacobi_kernel(
    const float* __restrict__ f, const float* __restrict__ u,
    float* __restrict__ out)
{
    int idx = blockIdx.x * 256 + threadIdx.x;         // < 32*128*128
    int b = idx >> 14;
    int p = idx & 16383;
    int y = p >> 7, x = p & 127;
    const float* ub = u + ((size_t)b << 14);
    float uc = ub[p];
    float s = ub[((y + 1) & 127) * NN + x] + ub[((y + 127) & 127) * NN + x]
            + ub[y * NN + ((x + 1) & 127)] + ub[y * NN + ((x + 127) & 127)];
    float lap = (s - 4.0f * uc) * INV_HH;
    out[idx] = uc + HH_OVER_4 * (f[idx] + lap);
}

// ---------------------------------------------------------------------------
// Fused ML step: res -> conv1(relu) -> conv2(relu, WMMA) -> conv3 -> u + corr.
// One workgroup = one (batch, 16x16 output tile). 256 threads = 8 wave32.
// ---------------------------------------------------------------------------
__global__ __launch_bounds__(256) void ml_step_kernel(
    const float* __restrict__ f, const float* __restrict__ uprev,
    const _Float16* __restrict__ w2f,
    const float* __restrict__ w1, const float* __restrict__ b1,
    const float* __restrict__ b2,
    const float* __restrict__ w3, const float* __restrict__ b3,
    float* __restrict__ out)
{
    __shared__ float uT[576];                       // 24x24 u tile (halo 4)
    __shared__ float resS[484];                     // 22x22 residual
    __shared__ float w1s[576], w3s[576];
    __shared__ float b1s[64], b2s[64];
    __shared__ __align__(32) _Float16 h1s[400 * 64]; // 20x20 x 64ch (f16)
    __shared__ __align__(32) _Float16 h2s[324 * 64]; // 18x18 x 64ch (f16)

    const int tid = threadIdx.x;
    const int b   = blockIdx.x >> 6;
    const int til = blockIdx.x & 63;
    const int ty  = til >> 3, tx = til & 7;
    const float* ub = uprev + (size_t)b * (NN * NN);
    const float* fb = f     + (size_t)b * (NN * NN);

    // ---- stage params + u tile ----
    for (int i = tid; i < 576; i += 256) { w1s[i] = w1[i]; w3s[i] = w3[i]; }
    if (tid < 64) { b1s[tid] = b1[tid]; b2s[tid] = b2[tid]; }
    for (int i = tid; i < 576; i += 256) {
        int y = i / 24, x = i % 24;
        int gy = (ty * TILE + y - 4) & (NN - 1);
        int gx = (tx * TILE + x - 4) & (NN - 1);
        uT[i] = ub[gy * NN + gx];
    }
    __syncthreads();

    // ---- residual 22x22: res = f + lap(u) ----
    for (int i = tid; i < 484; i += 256) {
        int y = i / 22, x = i % 22;
        float uc = uT[(y + 1) * 24 + (x + 1)];
        float s  = uT[y * 24 + (x + 1)] + uT[(y + 2) * 24 + (x + 1)]
                 + uT[(y + 1) * 24 + x] + uT[(y + 1) * 24 + (x + 2)];
        float lap = (s - 4.0f * uc) * INV_HH;
        int gy = (ty * TILE + y - 3) & (NN - 1);
        int gx = (tx * TILE + x - 3) & (NN - 1);
        resS[i] = fb[gy * NN + gx] + lap;
    }
    __syncthreads();

    // ---- conv1 (1->64) + relu, fp32 scalar; h1 is 20x20, channel-contiguous
    for (int i = tid; i < 400 * 64; i += 256) {
        int ch = i & 63, p = i >> 6;
        int y = p / 20, x = p % 20;
        float acc = b1s[ch];
        const float* wr = &w1s[ch * 9];
#pragma unroll
        for (int dy = 0; dy < 3; ++dy)
#pragma unroll
            for (int dx = 0; dx < 3; ++dx)
                acc = fmaf(wr[dy * 3 + dx], resS[(y + dy) * 22 + (x + dx)], acc);
        h1s[p * 64 + ch] = (_Float16)(acc > 0.0f ? acc : 0.0f);
    }
    __syncthreads();

    // ---- conv2 (64->64) + relu via WMMA f32<-f16; h2 is 18x18 = 324 px ----
    {
        const int wave = tid >> 5, lane = tid & 31;
        const int hi = (lane >> 4) & 1;              // upper half-wave
        for (int t = wave; t < 84; t += 8) {         // 4 ocTiles x 21 pixGroups
            const int ocT = t & 3, pg = t >> 2;
            int pix  = pg * 16 + (lane & 15);
            int pixc = pix < 324 ? pix : 323;
            int py = pixc / 18, px = pixc % 18;
            v8f acc0, acc1;
#pragma unroll
            for (int r = 0; r < 8; ++r) {
                acc0[r] = b2s[ocT * 16 + hi * 8 + r];
                acc1[r] = 0.0f;
            }
            for (int tap = 0; tap < 9; ++tap) {
                const int h1off = ((py + tap / 3) * 20 + (px + tap % 3)) * 64;
                const _Float16* ap = w2f + (size_t)((tap * 2) * 4 + ocT) * 512
                                         + lane * 16;
                v16h a0 = *(const v16h*)ap;                       // kc = 0
                v16h bb0 = *(const v16h*)&h1s[h1off + hi * 16];
                acc0 = __builtin_amdgcn_wmma_f32_16x16x32_f16(
                    false, a0, false, bb0, (short)0, acc0, false, false);
                v16h a1 = *(const v16h*)(ap + 4 * 512);           // kc = 1
                v16h bb1 = *(const v16h*)&h1s[h1off + 32 + hi * 16];
                acc1 = __builtin_amdgcn_wmma_f32_16x16x32_f16(
                    false, a1, false, bb1, (short)0, acc1, false, false);
            }
            if (pix < 324) {
                int base = pix * 64 + ocT * 16 + hi * 8;
#pragma unroll
                for (int r = 0; r < 8; ++r) {
                    float v = acc0[r] + acc1[r];
                    h2s[base + r] = (_Float16)(v > 0.0f ? v : 0.0f);
                }
            }
        }
    }
    __syncthreads();

    // ---- conv3 (64->1), add correction to u, store prediction ----
    {
        const int oy = tid >> 4, ox = tid & 15;
        float acc = b3[0];
#pragma unroll
        for (int dy = 0; dy < 3; ++dy)
#pragma unroll
            for (int dx = 0; dx < 3; ++dx) {
                const _Float16* hrow = &h2s[((oy + dy) * 18 + (ox + dx)) * 64];
                const int tap = dy * 3 + dx;
                for (int ic = 0; ic < 64; ++ic)
                    acc = fmaf(w3s[ic * 9 + tap], (float)hrow[ic], acc);
            }
        float val = uT[(oy + 4) * 24 + (ox + 4)] + acc;
        out[(size_t)b * (NN * NN) + (ty * TILE + oy) * NN + (tx * TILE + ox)] = val;
    }
}

// ---------------------------------------------------------------------------
// Routing scores: [8][32][2], one-hot by iteration schedule.
// ---------------------------------------------------------------------------
__global__ void scores_kernel(float* __restrict__ dst)
{
    int i = threadIdx.x;                  // 512
    int it = i >> 6;
    int k  = i & 1;
    int ml = ((it & 3) == 3) ? 1 : 0;
    dst[i] = (k == 1) ? (float)ml : (float)(1 - ml);
}

// ---------------------------------------------------------------------------
extern "C" void kernel_launch(void* const* d_in, const int* in_sizes, int n_in,
                              void* d_out, int out_size, void* d_ws, size_t ws_size,
                              hipStream_t stream)
{
    const float* f  = (const float*)d_in[0];
    const float* u0 = (const float*)d_in[1];
    const float* w1 = (const float*)d_in[2];
    const float* b1 = (const float*)d_in[3];
    const float* w2 = (const float*)d_in[4];
    const float* b2 = (const float*)d_in[5];
    const float* w3 = (const float*)d_in[6];
    const float* b3 = (const float*)d_in[7];
    float* pred = (float*)d_out;
    float* sc   = pred + (size_t)8 * 32 * NN * NN;
    _Float16* w2f = (_Float16*)d_ws;      // 73,728 bytes of workspace

    prep_w2_kernel<<<144, 256, 0, stream>>>(w2, w2f);

    const size_t slice = (size_t)32 * NN * NN;
    for (int it = 0; it < 8; ++it) {
        const float* uprev = (it == 0) ? u0 : pred + (size_t)(it - 1) * slice;
        float* out = pred + (size_t)it * slice;
        if ((it & 3) == 3) {
            ml_step_kernel<<<2048, 256, 0, stream>>>(f, uprev, w2f,
                                                     w1, b1, b2, w3, b3, out);
        } else {
            jacobi_kernel<<<2048, 256, 0, stream>>>(f, uprev, out);
        }
    }
    scores_kernel<<<1, 512, 0, stream>>>(sc);
}
